// Calibrated_Spectral_Mixer_TwoDomain_15831249453431
// MI455X (gfx1250) — matrix operand order repaired and checked by reading, not compiled, not run on hardware
//
#include <hip/hip_runtime.h>
#include <hip/hip_bf16.h>

// Problem constants (match reference)
#define BB 8
#define NN 8192
#define CC 128
#define HH 8
#define DHH 64
#define GG 32
#define CHUNKS 64
#define CH_ROWS (NN / CHUNKS) /* 128 */
#define LN_EPS 1e-5f

// LDS row strides (dwords): chosen so rows stay 16B-aligned (stride % 4 == 0)
// and fragment-gather lanes spread across banks.
#define XS_STRIDE 132  /* x chunk rows: 128 dw data + 4 pad */
#define AS_STRIDE 36   /* inv chunk rows: 32 dw data + 4 pad; 8*36 % 64 == 32 -> halves disjoint */
#define YS_STRIDE 132  /* y repack rows */
#define ZS_STRIDE 132  /* Z rows */

typedef __attribute__((ext_vector_type(16))) __bf16 v16bf;
typedef __attribute__((ext_vector_type(8)))  float  v8f;
typedef int vec4i __attribute__((vector_size(16)));   // pointee type of async-LDS builtin

// A-matrix (16x32 bf16) K index for element j of the lane's v16bf,
// per ISA 7.12.2: lanes0-15: V0..3 hold K0..7, V4..7 hold K16..23;
// lanes16-31: K8..15 and K24..31.
static __device__ __forceinline__ int kofA(int j, int half) {
  return (j & 7) + ((j >> 3) << 4) + (half << 3);
}

// ---- CDNA5 async global->LDS staging (ASYNCcnt path), with safe fallback ----
#if defined(__has_builtin)
#if __has_builtin(__builtin_amdgcn_global_load_async_to_lds_b128)
#define HAVE_ASYNC_LDS 1
#endif
#endif
#ifndef HAVE_ASYNC_LDS
#define HAVE_ASYNC_LDS 0
#endif

static __device__ __forceinline__ void copy16_to_lds(const float* g, float* l) {
#if HAVE_ASYNC_LDS
  __builtin_amdgcn_global_load_async_to_lds_b128(
      (__attribute__((address_space(1))) vec4i*)g,
      (__attribute__((address_space(3))) vec4i*)l, 0, 0);
#else
  *(float4*)l = *(const float4*)g;
#endif
}

static __device__ __forceinline__ void staging_wait() {
#if HAVE_ASYNC_LDS
#if __has_builtin(__builtin_amdgcn_s_wait_asynccnt)
  __builtin_amdgcn_s_wait_asynccnt(0);
#else
  asm volatile("s_wait_asynccnt 0" ::: "memory");
#endif
#endif
}

// ---------------------------------------------------------------------------
// K1: T-partials[b,chunk] (32x128) = inv_in(chunk rows)^T @ x[b](chunk rows)
// x and inv chunks are staged to LDS with coalesced B128 (async if available),
// fragments gathered from LDS, accumulated with v_wmma_f32_16x16x32_bf16.
// ---------------------------------------------------------------------------
__global__ __launch_bounds__(256)
void k1_encode(const float* __restrict__ x, const float* __restrict__ inv_in,
               float* __restrict__ partials) {
  const int chunk = blockIdx.x;
  const int b     = blockIdx.y;
  const int n0    = chunk * CH_ROWS;
  const int tid   = threadIdx.x;
  const int wave  = tid >> 5;
  const int lane  = tid & 31;
  const int half  = lane >> 4;
  const int lr    = lane & 15;
  const int mtile = wave >> 2;        // 2 tiles over G=32
  const int npair = wave & 3;         // 4 pairs over C=128 (8 ntiles)
  const int gbase = mtile * 16;
  const int c0    = (2 * npair) * 16;
  const int c1    = c0 + 16;
  const float* xb   = x + (size_t)b * NN * CC;
  const float* invc = inv_in;         // head 0: bases identical across H

  __shared__ float Xs[CH_ROWS * XS_STRIDE]; // 66 KB, layout [k][c]
  __shared__ float As[CH_ROWS * AS_STRIDE]; // 18 KB, layout [k][g]

  // Stage x chunk: wave w copies rows 16w..16w+15; one B128 per lane per row.
  for (int rr = 0; rr < 16; ++rr) {
    const int k = wave * 16 + rr;
    copy16_to_lds(&xb[(size_t)(n0 + k) * CC + 4 * lane], &Xs[k * XS_STRIDE + 4 * lane]);
  }
  // Stage inv chunk: 4 rows per instruction (32 lanes x 16B = 4 x 32 floats).
  for (int it = 0; it < 4; ++it) {
    const int k  = wave * 16 + it * 4 + (lane >> 3);
    const int c4 = (lane & 7) * 4;
    copy16_to_lds(&invc[(size_t)(n0 + k) * GG + c4], &As[k * AS_STRIDE + c4]);
  }
  staging_wait();
  __syncthreads();

  v8f acc0 = {};
  v8f acc1 = {};
  for (int kb = 0; kb < CH_ROWS / 32; ++kb) {
    const int kbase = kb * 32;
    v16bf a, b0, b1;
#pragma unroll
    for (int j = 0; j < 16; ++j) {
      const int k = kofA(j, half);
      a[j] = (__bf16)As[(kbase + k) * AS_STRIDE + gbase + lr];
    }
#pragma unroll
    for (int j = 0; j < 16; ++j) {
      const int k = kbase + kofA(j, half);
      b0[j] = (__bf16)Xs[k * XS_STRIDE + c0 + lr];
      b1[j] = (__bf16)Xs[k * XS_STRIDE + c1 + lr];
    }
    acc0 = __builtin_amdgcn_wmma_f32_16x16x32_bf16(false, a, false, b0, (short)0, acc0, false, false);
    acc1 = __builtin_amdgcn_wmma_f32_16x16x32_bf16(false, a, false, b1, (short)0, acc1, false, false);
  }
  float* P = partials + (size_t)(b * CHUNKS + chunk) * GG * CC;
#pragma unroll
  for (int r = 0; r < 8; ++r) {
    const int g = gbase + r + half * 8;
    P[(size_t)g * CC + c0 + lr] = acc0[r];
    P[(size_t)g * CC + c1 + lr] = acc1[r];
  }
}

// ---------------------------------------------------------------------------
// K2pre: T[b] = sum over chunks of partials (fixed order, float4 b128 path);
// block 0 also computes s[g] = sum_n inv_in[n,g] (for the b_in bias fold).
// ---------------------------------------------------------------------------
__global__ __launch_bounds__(256)
void k2_reduce(const float* __restrict__ partials, const float* __restrict__ inv_in,
               float* __restrict__ T, float* __restrict__ s) {
  const int b   = blockIdx.x;
  const int tid = threadIdx.x;
  const float4* pb = (const float4*)(partials + (size_t)b * CHUNKS * GG * CC);
  float4* tb = (float4*)(T + (size_t)b * GG * CC);
#pragma unroll
  for (int i = 0; i < 4; ++i) {
    const int idx = tid + 256 * i;          // 1024 float4 = 32x128 dwords
    float ax = 0.f, ay = 0.f, az = 0.f, aw = 0.f;
    for (int ch = 0; ch < CHUNKS; ++ch) {
      const float4 v = pb[(size_t)ch * (GG * CC / 4) + idx];
      ax += v.x; ay += v.y; az += v.z; aw += v.w;
    }
    float4 o; o.x = ax; o.y = ay; o.z = az; o.w = aw;
    tb[idx] = o;
  }
  if (b == 0) {
    __shared__ float red[256];
    const int g = tid & 31, part = tid >> 5;          // 8 partial sums per g
    float acc = 0.f;
    const int span = NN / 8;
    for (int n = part * span; n < (part + 1) * span; ++n)
      acc += inv_in[(size_t)n * GG + g];
    red[tid] = acc;
    __syncthreads();
    if (tid < GG) {
      float t = 0.f;
      for (int p2 = 0; p2 < 8; ++p2) t += red[p2 * 32 + tid];
      s[tid] = t;
    }
  }
}

// ---------------------------------------------------------------------------
// K2a (grid B*H): per (b,h): spec_h(32x64) = T[b] @ W_in[:,h slice] + s*b_in,
// LayerNorm over the 32x64 tile, apply ln_g/ln_b, then per-head mlp_w mix.
// ---------------------------------------------------------------------------
__global__ __launch_bounds__(256)
void k2_mid(const float* __restrict__ T, const float* __restrict__ s,
            const float* __restrict__ W_in, const float* __restrict__ b_in,
            const float* __restrict__ ln_g, const float* __restrict__ ln_b,
            const float* __restrict__ mlp_w, float* __restrict__ specm) {
  const int b   = blockIdx.x >> 3;
  const int h   = blockIdx.x & 7;
  const int tid = threadIdx.x;
  __shared__ float Tl[GG * CC];     // 16 KB
  __shared__ float sp[GG * DHH];    // 8 KB
  __shared__ float red[256];
  __shared__ float stats[2];

  for (int i = tid; i < GG * CC; i += 256) Tl[i] = T[(size_t)b * GG * CC + i];
  __syncthreads();

  float v[8];
  float psum = 0.f, psq = 0.f;
#pragma unroll
  for (int i = 0; i < 8; ++i) {
    const int e = tid + 256 * i;    // 2048 = 32x64
    const int g = e >> 6, d = e & 63;
    float acc = s[g] * b_in[h * DHH + d];
    const float* w = W_in + h * DHH + d;
    for (int c = 0; c < CC; ++c) acc += Tl[g * CC + c] * w[(size_t)c * (HH * DHH)];
    v[i] = acc;
    psum += acc;
    psq  += acc * acc;
  }
  red[tid] = psum;
  __syncthreads();
  for (int off = 128; off > 0; off >>= 1) { if (tid < off) red[tid] += red[tid + off]; __syncthreads(); }
  if (tid == 0) stats[0] = red[0] / (float)(GG * DHH);
  __syncthreads();
  red[tid] = psq;
  __syncthreads();
  for (int off = 128; off > 0; off >>= 1) { if (tid < off) red[tid] += red[tid + off]; __syncthreads(); }
  if (tid == 0) {
    const float mu  = stats[0];
    const float var = red[0] / (float)(GG * DHH) - mu * mu;
    stats[1] = rsqrtf(var + LN_EPS);
  }
  __syncthreads();
  const float mu = stats[0], rs = stats[1];
#pragma unroll
  for (int i = 0; i < 8; ++i) {
    const int e = tid + 256 * i;
    const int g = e >> 6, d = e & 63;
    sp[e] = (v[i] - mu) * rs * ln_g[g * DHH + d] + ln_b[g * DHH + d];
  }
  __syncthreads();
#pragma unroll
  for (int i = 0; i < 8; ++i) {
    const int e = tid + 256 * i;
    const int g = e >> 6, o = e & 63;
    float acc = 0.f;
    for (int ii = 0; ii < DHH; ++ii) acc += sp[g * DHH + ii] * mlp_w[ii * DHH + o];
    specm[((size_t)b * GG + g) * (HH * DHH) + h * DHH + o] = acc;
  }
}

// ---------------------------------------------------------------------------
// K2b (grid B*8): Z[b](32x128) = specm[b](32x512) @ W_out(512x128)
// ---------------------------------------------------------------------------
__global__ __launch_bounds__(256)
void k2_out(const float* __restrict__ specm, const float* __restrict__ W_out,
            float* __restrict__ Z) {
  const int b   = blockIdx.x >> 3;
  const int ct  = blockIdx.x & 7;
  const int tid = threadIdx.x;
#pragma unroll
  for (int i = 0; i < 2; ++i) {
    const int e = tid + 256 * i;               // 512 = 32x16
    const int g = e >> 4;
    const int c = ct * 16 + (e & 15);
    float acc = 0.f;
    const float* sm = specm + ((size_t)b * GG + g) * (HH * DHH);
    for (int hd = 0; hd < HH * DHH; ++hd) acc += sm[hd] * W_out[(size_t)hd * CC + c];
    Z[((size_t)b * GG + g) * CC + c] = acc;
  }
}

// ---------------------------------------------------------------------------
// K3: y[b] (8192x128) = inv_out(8192x32) @ Z[b](32x128) + b_out
// One WMMA (K=32) per 16x16 output tile. A loaded as float4 (b128), Z staged
// to LDS; results repacked through LDS so y is written with full-row B128.
// ---------------------------------------------------------------------------
__global__ __launch_bounds__(256)
void k3_decode(const float* __restrict__ inv_out, const float* __restrict__ Z,
               const float* __restrict__ b_out, float* __restrict__ y) {
  const int b    = blockIdx.y;
  const int n0   = blockIdx.x * 128;
  const int tid  = threadIdx.x;
  const int wave = tid >> 5;                   // wave = mtile over 128 rows
  const int lane = tid & 31;
  const int half = lane >> 4;
  const int lr   = lane & 15;
  const int row  = n0 + wave * 16 + lr;

  __shared__ float Zs[GG * ZS_STRIDE];          // ~17 KB, layout [g][c]
  __shared__ float Ys[8 * 16 * YS_STRIDE];      // ~66 KB, per-wave repack

  const float* Zb = Z + (size_t)b * GG * CC;
  float* yb = y + (size_t)b * NN * CC;

  // Stage Z (32x128 f32): wave w copies rows 4w..4w+3, one full row per copy op.
  for (int it = 0; it < 4; ++it) {
    const int g = wave * 4 + it;
    copy16_to_lds(&Zb[(size_t)g * CC + 4 * lane], &Zs[g * ZS_STRIDE + 4 * lane]);
  }
  staging_wait();
  __syncthreads();

  // A fragment: two contiguous 8-float runs per lane -> b128 loads.
  const float* arow = inv_out + (size_t)row * GG + half * 8;
  const float4 a0 = *(const float4*)(arow + 0);
  const float4 a1 = *(const float4*)(arow + 4);
  const float4 a2 = *(const float4*)(arow + 16);
  const float4 a3 = *(const float4*)(arow + 20);
  const float af[16] = {a0.x, a0.y, a0.z, a0.w, a1.x, a1.y, a1.z, a1.w,
                        a2.x, a2.y, a2.z, a2.w, a3.x, a3.y, a3.z, a3.w};
  v16bf a;
#pragma unroll
  for (int j = 0; j < 16; ++j) a[j] = (__bf16)af[j];

  float* yw = &Ys[wave * 16 * YS_STRIDE];
  for (int nt = 0; nt < 8; ++nt) {
    const int cb = nt * 16;
    const float bias = b_out[cb + lr];
    v8f acc = {bias, bias, bias, bias, bias, bias, bias, bias};
    v16bf bm;
#pragma unroll
    for (int j = 0; j < 16; ++j)
      bm[j] = (__bf16)Zs[(size_t)kofA(j, half) * ZS_STRIDE + cb + lr];
    acc = __builtin_amdgcn_wmma_f32_16x16x32_bf16(false, a, false, bm, (short)0, acc, false, false);
#pragma unroll
    for (int r = 0; r < 8; ++r)
      yw[(r + half * 8) * YS_STRIDE + cb + lr] = acc[r];
  }

  // Wave-local repack: read full 128-float rows and store y with B128.
#pragma unroll 4
  for (int i = 0; i < 16; ++i) {
    const float4 val = *(const float4*)&yw[i * YS_STRIDE + 4 * lane];
    *(float4*)&yb[(size_t)(n0 + wave * 16 + i) * CC + 4 * lane] = val;
  }
}

// ---------------------------------------------------------------------------
extern "C" void kernel_launch(void* const* d_in, const int* in_sizes, int n_in,
                              void* d_out, int out_size, void* d_ws, size_t ws_size,
                              hipStream_t stream) {
  const float* x       = (const float*)d_in[0];
  const float* W_in    = (const float*)d_in[1];
  const float* b_in    = (const float*)d_in[2];
  const float* mlp_w   = (const float*)d_in[3];
  const float* ln_g    = (const float*)d_in[4];
  const float* ln_b    = (const float*)d_in[5];
  const float* W_out   = (const float*)d_in[6];
  const float* b_out   = (const float*)d_in[7];
  const float* inv_in  = (const float*)d_in[8];   // (H,N,G), heads identical
  const float* inv_out = (const float*)d_in[9];   // (H,N,G), heads identical
  float* y  = (float*)d_out;
  float* ws = (float*)d_ws;

  // Workspace layout (floats): ~9.2 MB total
  float* partials = ws;                                           // B*CHUNKS*G*C
  float* T        = partials + (size_t)BB * CHUNKS * GG * CC;     // B*G*C
  float* s        = T + (size_t)BB * GG * CC;                     // G
  float* specm    = s + GG;                                       // B*G*(H*DH)
  float* Z        = specm + (size_t)BB * GG * HH * DHH;           // B*G*C

  k1_encode<<<dim3(CHUNKS, BB), 256, 0, stream>>>(x, inv_in, partials);
  k2_reduce<<<dim3(BB), 256, 0, stream>>>(partials, inv_in, T, s);
  k2_mid<<<dim3(BB * HH), 256, 0, stream>>>(T, s, W_in, b_in, ln_g, ln_b, mlp_w, specm);
  k2_out<<<dim3(BB * 8), 256, 0, stream>>>(specm, W_out, Z);
  k3_decode<<<dim3(NN / 128, BB), 256, 0, stream>>>(inv_out, Z, b_out, y);
}
